// FramePredictor_20950850470152
// MI455X (gfx1250) — matrix operand order repaired and checked by reading, not compiled, not run on hardware
//
#include <hip/hip_runtime.h>
#include <hip/hip_bf16.h>
#include <math.h>

// ---------------- problem constants ----------------
constexpr int Bn   = 4;
constexpr int Tn   = 1024;
constexpr int Dn   = 1024;
constexpr int Hn   = 16;
constexpr int KVHn = 4;
constexpr int HDn  = 64;
constexpr int DFFn = 4096;
constexpr int WINn = 256;
constexpr int Mrows = Bn * Tn;          // 4096

// ---------------- vector types ----------------
typedef __attribute__((ext_vector_type(8)))  float          v8f;
typedef __attribute__((ext_vector_type(8)))  unsigned short v8u;
typedef __attribute__((ext_vector_type(16))) unsigned short v16u;
typedef __attribute__((ext_vector_type(16))) __bf16         v16bf;
typedef __attribute__((ext_vector_type(4)))  unsigned int   v4ui;
typedef __attribute__((ext_vector_type(8)))  int            v8i;
typedef __attribute__((ext_vector_type(4)))  int            v4i;

// Tensor Data Mover availability (device pass only; fallback keeps compile safe)
#if defined(__HIP_DEVICE_COMPILE__) && \
    __has_builtin(__builtin_amdgcn_tensor_load_to_lds) && \
    __has_builtin(__builtin_amdgcn_s_wait_tensorcnt)
#define USE_TDM 1
#else
#define USE_TDM 0
#endif

// ---------------- helpers ----------------
__device__ __forceinline__ unsigned short f2bf(float f) {
    unsigned int u = __float_as_uint(f);
    u += 0x7fffu + ((u >> 16) & 1u);       // round-to-nearest-even
    return (unsigned short)(u >> 16);
}

__device__ __forceinline__ v8f wmma_bf16(v16bf a, v16bf b, v8f c) {
    // v_wmma_f32_16x16x32_bf16 : D = A(16x32) * B(32x16) + C(16x16 f32)
    return __builtin_amdgcn_wmma_f32_16x16x32_bf16(
        false, a, false, b, (short)0, c, false, false);
}

// A-fragment (16x32 bf16): lane holds row M=lane&15; u[0..15] = K kg*16+0..15 (contiguous, matches B).
__device__ __forceinline__ v16bf load_a_frag(const unsigned short* p) {
    return __builtin_bit_cast(v16bf, *(const v16u*)p);
}

// B-fragment (32x16 bf16): lane = column, K contiguous per half-wave -> one 32B load.
__device__ __forceinline__ v16bf load_b_frag(const unsigned short* p) {
    return __builtin_bit_cast(v16bf, *(const v16u*)p);
}

#if USE_TDM
// Issue one TDM 2D tile load: gsrc is tile start inside a [tensor_rows x Kstride]
// bf16 tensor; loads (rows x cols) elements to LDS offset lds_off, row-major packed.
// D# packing per CDNA5 ISA ch.8 (group0: count/lds/global/type, group1: dims+strides).
__device__ __forceinline__ void tdm_load_2d(const unsigned short* gsrc,
                                            unsigned lds_off,
                                            unsigned rows, unsigned cols,
                                            unsigned Kstride, unsigned tensor_rows) {
    unsigned long long ga = (unsigned long long)(size_t)gsrc;
    v4ui g0;
    g0[0] = 1u;                                            // count=1, user mode
    g0[1] = lds_off;                                       // lds_addr
    g0[2] = (unsigned)(ga & 0xffffffffu);                  // global_addr[31:0]
    g0[3] = (unsigned)((ga >> 32) & 0x01ffffffu)           // global_addr[56:32]
          | (2u << 30);                                    // type = 2 ("image")
    v8i g1;
    g1[0] = (int)(1u << 16);                               // data_size=1 (2 bytes)
    g1[1] = (int)((Kstride & 0xffffu) << 16);              // tensor_dim0[15:0]
    g1[2] = (int)((Kstride >> 16) | ((tensor_rows & 0xffffu) << 16));
    g1[3] = (int)((tensor_rows >> 16) | (cols << 16));     // tile_dim0 = cols
    g1[4] = (int)rows;                                     // tile_dim1; tile_dim2=0
    g1[5] = (int)Kstride;                                  // tensor_dim0_stride[31:0]
    g1[6] = 0;                                             // stride hi / dim1_stride
    g1[7] = 0;
    v4i z = {0, 0, 0, 0};
#if defined(__clang_major__) && (__clang_major__ >= 23)
    v8i z8 = {};
    __builtin_amdgcn_tensor_load_to_lds(g0, g1, z, z, z8, 0);
#else
    __builtin_amdgcn_tensor_load_to_lds(g0, g1, z, z, 0);
#endif
}
#endif

// ---------------- weight convert + transpose: W[K,N] f32 -> Wt[N,K] bf16 ----
__global__ __launch_bounds__(256)
void wconv_kernel(const float* __restrict__ W, unsigned short* __restrict__ Wt,
                  int K, int N) {
    int idx = blockIdx.x * 256 + threadIdx.x;     // enumerates output [N][K]
    if (idx >= K * N) return;
    int k = idx % K;
    int n = idx / K;
    Wt[idx] = f2bf(W[(size_t)k * N + n]);
}

// ---------------- layernorm -> bf16 ----------------
__global__ __launch_bounds__(256)
void ln_kernel(const float* __restrict__ x, const float* __restrict__ g,
               const float* __restrict__ b, unsigned short* __restrict__ out,
               int D) {
    int row = blockIdx.x;
    const float* xr = x + (size_t)row * D;
    __shared__ float s1[256], s2[256];
    float sum = 0.f, sq = 0.f;
    for (int i = threadIdx.x; i < D; i += 256) {
        float v = xr[i];
        sum += v; sq += v * v;
    }
    s1[threadIdx.x] = sum; s2[threadIdx.x] = sq;
    __syncthreads();
#pragma unroll
    for (int s = 128; s > 0; s >>= 1) {
        if (threadIdx.x < s) {
            s1[threadIdx.x] += s1[threadIdx.x + s];
            s2[threadIdx.x] += s2[threadIdx.x + s];
        }
        __syncthreads();
    }
    float mu  = s1[0] / (float)D;
    float var = s2[0] / (float)D - mu * mu;
    float rs  = rsqrtf(var + 1e-5f);
    unsigned short* o = out + (size_t)row * D;
    for (int i = threadIdx.x; i < D; i += 256)
        o[i] = f2bf((xr[i] - mu) * rs * g[i] + b[i]);
}

// ---------------- TDM-staged, double-buffered WMMA GEMM ----------------
// Block = 8 waves computes a 128x64 tile of C = A_bf16[M,K] @ Wt_bf16[N,K]^T + bias.
// Software pipeline: wave 0 issues the NEXT K-chunk's two tensor_load_to_lds DMAs
// right after the barrier, all waves run WMMA from the CURRENT LDS buffer, and the
// s_wait_tensorcnt at the next iteration's top overlaps DMA with matrix compute.
// Epilogue MODE: 0 f32 out, 1 f32 out + residual, 2 exact-GELU -> bf16 out.
template <int MODE>
__global__ __launch_bounds__(256)
void gemm_kernel(const unsigned short* __restrict__ A,
                 const unsigned short* __restrict__ Bt,
                 const float* __restrict__ bias,
                 const float* __restrict__ resid,
                 float* __restrict__ outF,
                 unsigned short* __restrict__ outB,
                 int M, int N, int K) {
    __shared__ unsigned short As[2][128 * 64];    // 2 x 16 KB
    __shared__ unsigned short Bs[2][64 * 64];     // 2 x  8 KB
    const int wid  = threadIdx.x >> 5;
    const int lane = threadIdx.x & 31;
    const int ln15 = lane & 15;
    const int kg   = lane >> 4;
    const int ntN  = N >> 6;
    const int mb   = blockIdx.x / ntN;
    const int nt   = blockIdx.x % ntN;
    const int m0   = mb * 128;

    const unsigned short* Abase = A  + (size_t)m0 * K;
    const unsigned short* Bbase = Bt + (size_t)(nt * 64) * K;

    v8f c0 = {}, c1 = {}, c2 = {}, c3 = {};

#if USE_TDM
    if (wid == 0) {                            // prologue: fill buffer 0
        tdm_load_2d(Abase, (unsigned)(size_t)&As[0][0], 128, 64,
                    (unsigned)K, (unsigned)(M - m0));
        tdm_load_2d(Bbase, (unsigned)(size_t)&Bs[0][0], 64, 64,
                    (unsigned)K, (unsigned)N);
    }
    int buf = 0;
    for (int kk = 0; kk < K; kk += 64) {
        if (wid == 0)
            __builtin_amdgcn_s_wait_tensorcnt(0);   // current buffer landed
        __syncthreads();
        if (wid == 0 && kk + 64 < K) {              // prefetch next chunk
            tdm_load_2d(Abase + kk + 64, (unsigned)(size_t)&As[buf ^ 1][0],
                        128, 64, (unsigned)K, (unsigned)(M - m0));
            tdm_load_2d(Bbase + kk + 64, (unsigned)(size_t)&Bs[buf ^ 1][0],
                        64, 64, (unsigned)K, (unsigned)N);
        }
        const unsigned short* Ab = &As[buf][0];
        const unsigned short* Bb = &Bs[buf][0];
#pragma unroll
        for (int k2 = 0; k2 < 64; k2 += 32) {
            v16bf a = load_a_frag(&Ab[(wid * 16 + ln15) * 64 + k2 + kg * 16]);
            c0 = wmma_bf16(a, load_b_frag(&Bb[( 0 + ln15) * 64 + k2 + kg * 16]), c0);
            c1 = wmma_bf16(a, load_b_frag(&Bb[(16 + ln15) * 64 + k2 + kg * 16]), c1);
            c2 = wmma_bf16(a, load_b_frag(&Bb[(32 + ln15) * 64 + k2 + kg * 16]), c2);
            c3 = wmma_bf16(a, load_b_frag(&Bb[(48 + ln15) * 64 + k2 + kg * 16]), c3);
        }
        __syncthreads();
        buf ^= 1;
    }
#else
    for (int kk = 0; kk < K; kk += 64) {
        __syncthreads();
        for (int i = threadIdx.x; i < (128 * 64) / 8; i += 256) {
            int row = i >> 3, c8 = (i & 7) * 8;
            *(v8u*)&As[0][row * 64 + c8] =
                *(const v8u*)(Abase + (size_t)row * K + kk + c8);
        }
        for (int i = threadIdx.x; i < (64 * 64) / 8; i += 256) {
            int row = i >> 3, c8 = (i & 7) * 8;
            *(v8u*)&Bs[0][row * 64 + c8] =
                *(const v8u*)(Bbase + (size_t)row * K + kk + c8);
        }
        __syncthreads();
#pragma unroll
        for (int k2 = 0; k2 < 64; k2 += 32) {
            v16bf a = load_a_frag(&As[0][(wid * 16 + ln15) * 64 + k2 + kg * 16]);
            c0 = wmma_bf16(a, load_b_frag(&Bs[0][( 0 + ln15) * 64 + k2 + kg * 16]), c0);
            c1 = wmma_bf16(a, load_b_frag(&Bs[0][(16 + ln15) * 64 + k2 + kg * 16]), c1);
            c2 = wmma_bf16(a, load_b_frag(&Bs[0][(32 + ln15) * 64 + k2 + kg * 16]), c2);
            c3 = wmma_bf16(a, load_b_frag(&Bs[0][(48 + ln15) * 64 + k2 + kg * 16]), c3);
        }
    }
#endif

    // C layout: lane holds column n = sub*16+ln15, rows m = 8*kg + r
#pragma unroll
    for (int s = 0; s < 4; ++s) {
        v8f c = (s == 0) ? c0 : (s == 1) ? c1 : (s == 2) ? c2 : c3;
        int col = nt * 64 + s * 16 + ln15;
        float bi = bias[col];
#pragma unroll
        for (int r = 0; r < 8; ++r) {
            int rowo = m0 + wid * 16 + 8 * kg + r;
            float v = c[r] + bi;
            size_t oidx = (size_t)rowo * N + col;
            if (MODE == 0) {
                outF[oidx] = v;
            } else if (MODE == 1) {
                outF[oidx] = v + resid[oidx];
            } else {
                float gl = 0.5f * v * (1.0f + erff(v * 0.70710678118654752f));
                outB[oidx] = f2bf(gl);
            }
        }
    }
}

// ---------------- RoPE f32 [B,T,nh,HD] -> bf16 [B,nh,T,HD] ----------------
__global__ __launch_bounds__(256)
void rope_kernel(const float* __restrict__ in, unsigned short* __restrict__ out,
                 int nh) {
    int idx = blockIdx.x * 256 + threadIdx.x;      // (b, t, h, d<32)
    int d = idx & 31;
    int h = (idx >> 5) % nh;
    int t = (idx >> 5) / nh % Tn;
    int b = idx / (32 * nh * Tn);
    if (b >= Bn) return;
    const float* p = in + ((size_t)(b * Tn + t) * nh + h) * HDn;
    float x1 = p[d], x2 = p[d + 32];
    float fr = (float)t * expf(-(float)d * (9.2103403719761836f / 32.0f));
    float cs = cosf(fr), sn = sinf(fr);
    unsigned short* o = out + ((size_t)(b * nh + h) * Tn + t) * HDn;
    o[d]      = f2bf(x1 * cs - x2 * sn);
    o[d + 32] = f2bf(x2 * cs + x1 * sn);
}

// ---------------- V f32 [B,T,KVH,HD] -> bf16 transposed [B,KVH,HD,T] --------
__global__ __launch_bounds__(256)
void vtrans_kernel(const float* __restrict__ v, unsigned short* __restrict__ vt) {
    int idx = blockIdx.x * 256 + threadIdx.x;      // enumerates output
    int t  = idx & (Tn - 1);
    int d  = (idx / Tn) % HDn;
    int kv = (idx / (Tn * HDn)) % KVHn;
    int b  = idx / (Tn * HDn * KVHn);
    if (b >= Bn) return;
    vt[idx] = f2bf(v[((size_t)(b * Tn + t) * KVHn + kv) * HDn + d]);
}

// ---------------- flash sliding-window GQA attention, all-WMMA --------------
__global__ __launch_bounds__(32)
void attn_kernel(const unsigned short* __restrict__ Qbf,
                 const unsigned short* __restrict__ Kbf,
                 const unsigned short* __restrict__ Vt,
                 unsigned short* __restrict__ Obf) {
    const int lane = threadIdx.x & 31;
    const int ln15 = lane & 15;
    const int kg   = lane >> 4;
    const int qb   = blockIdx.x % (Tn / 16);
    const int h    = (blockIdx.x / (Tn / 16)) % Hn;
    const int b    = blockIdx.x / ((Tn / 16) * Hn);
    const int kv   = h / (Hn / KVHn);
    const int q0   = qb * 16;

    __shared__ unsigned short pbuf[16 * 32];   // P tile, [query][key] bf16
    __shared__ float bc[16];                   // per-query broadcast slot

    const unsigned short* Qb = Qbf + ((size_t)(b * Hn + h) * Tn + q0 + ln15) * HDn;
    v16bf aq0 = load_a_frag(Qb + kg * 16);
    v16bf aq1 = load_a_frag(Qb + 32 + kg * 16);

    const unsigned short* Kb = Kbf + (size_t)(b * KVHn + kv) * Tn * HDn;
    const unsigned short* Vb = Vt  + (size_t)(b * KVHn + kv) * HDn * Tn;

    v8f o0 = {}, o1 = {}, o2 = {}, o3 = {};
    float mrun[8], lrun[8];
#pragma unroll
    for (int r = 0; r < 8; ++r) { mrun[r] = -1e30f; lrun[r] = 0.f; }

    int kb0 = q0 - (WINn - 1);
    if (kb0 < 0) kb0 = 0;
    kb0 &= ~31;

    for (int kb = kb0; kb <= q0 + 15; kb += 32) {
        const int key0 = kb + ln15;
        const int key1 = kb + 16 + ln15;
        const unsigned short* K0 = Kb + (size_t)key0 * HDn + kg * 16;
        const unsigned short* K1 = Kb + (size_t)key1 * HDn + kg * 16;
        v8f s0 = {}, s1 = {};
        s0 = wmma_bf16(aq0, load_b_frag(K0),      s0);
        s0 = wmma_bf16(aq1, load_b_frag(K0 + 32), s0);
        s1 = wmma_bf16(aq0, load_b_frag(K1),      s1);
        s1 = wmma_bf16(aq1, load_b_frag(K1 + 32), s1);

        float p0[8], p1[8], rm[8];
#pragma unroll
        for (int r = 0; r < 8; ++r) {
            int q = q0 + 8 * kg + r;
            bool ok0 = (key0 <= q) && (q - key0 < WINn);
            bool ok1 = (key1 <= q) && (q - key1 < WINn);
            p0[r] = ok0 ? s0[r] * 0.125f : -1e30f;
            p1[r] = ok1 ? s1[r] * 0.125f : -1e30f;
            rm[r] = fmaxf(p0[r], p1[r]);
        }
#pragma unroll
        for (int mk = 1; mk <= 8; mk <<= 1)
#pragma unroll
            for (int r = 0; r < 8; ++r)
                rm[r] = fmaxf(rm[r], __shfl_xor(rm[r], mk, 32));

        float al[8], ls[8];
#pragma unroll
        for (int r = 0; r < 8; ++r) {
            float mn = fmaxf(mrun[r], rm[r]);
            al[r] = expf(mrun[r] - mn);
            mrun[r] = mn;
            p0[r] = (p0[r] > -1e29f) ? expf(p0[r] - mn) : 0.f;
            p1[r] = (p1[r] > -1e29f) ? expf(p1[r] - mn) : 0.f;
            ls[r] = p0[r] + p1[r];
        }
#pragma unroll
        for (int mk = 1; mk <= 8; mk <<= 1)
#pragma unroll
            for (int r = 0; r < 8; ++r)
                ls[r] += __shfl_xor(ls[r], mk, 32);
#pragma unroll
        for (int r = 0; r < 8; ++r)
            lrun[r] = lrun[r] * al[r] + ls[r];

        if (ln15 == 0)
#pragma unroll
            for (int r = 0; r < 8; ++r) bc[kg * 8 + r] = al[r];
#pragma unroll
        for (int r = 0; r < 8; ++r) {
            pbuf[(kg * 8 + r) * 32 + ln15]      = f2bf(p0[r]);
            pbuf[(kg * 8 + r) * 32 + 16 + ln15] = f2bf(p1[r]);
        }
        __syncthreads();

        float sc = bc[ln15];
#pragma unroll
        for (int r = 0; r < 8; ++r) {
            o0[r] *= sc; o1[r] *= sc; o2[r] *= sc; o3[r] *= sc;
        }

        v16bf pb = load_b_frag(pbuf + ln15 * 32 + kg * 16);

#pragma unroll
        for (int dt = 0; dt < 4; ++dt) {
            const unsigned short* Vp =
                Vb + (size_t)(dt * 16 + ln15) * Tn + kb + kg * 16;
            v16bf av = load_a_frag(Vp);
            if (dt == 0) o0 = wmma_bf16(av, pb, o0);
            else if (dt == 1) o1 = wmma_bf16(av, pb, o1);
            else if (dt == 2) o2 = wmma_bf16(av, pb, o2);
            else o3 = wmma_bf16(av, pb, o3);
        }
        __syncthreads();
    }

    if (ln15 == 0)
#pragma unroll
        for (int r = 0; r < 8; ++r) bc[kg * 8 + r] = 1.0f / lrun[r];
    __syncthreads();
    float inv = bc[ln15];
    unsigned short* Op =
        Obf + ((size_t)(b * Tn) + q0 + ln15) * Dn + h * HDn;
#pragma unroll
    for (int r = 0; r < 8; ++r) {
        Op[ 0 + kg * 8 + r] = f2bf(o0[r] * inv);
        Op[16 + kg * 8 + r] = f2bf(o1[r] * inv);
        Op[32 + kg * 8 + r] = f2bf(o2[r] * inv);
        Op[48 + kg * 8 + r] = f2bf(o3[r] * inv);
    }
}

// ---------------- launch ----------------
extern "C" void kernel_launch(void* const* d_in, const int* in_sizes, int n_in,
                              void* d_out, int out_size, void* d_ws, size_t ws_size,
                              hipStream_t stream) {
    (void)in_sizes; (void)n_in; (void)out_size; (void)ws_size;
    const float* x   = (const float*)d_in[0];
    const float* Wq  = (const float*)d_in[1];
    const float* bq  = (const float*)d_in[2];
    const float* Wk  = (const float*)d_in[3];
    const float* bk  = (const float*)d_in[4];
    const float* Wv  = (const float*)d_in[5];
    const float* bv  = (const float*)d_in[6];
    const float* Wo  = (const float*)d_in[7];
    const float* bo  = (const float*)d_in[8];
    const float* g1  = (const float*)d_in[9];
    const float* b1  = (const float*)d_in[10];
    const float* Wf1 = (const float*)d_in[11];
    const float* bf1 = (const float*)d_in[12];
    const float* Wf2 = (const float*)d_in[13];
    const float* bf2 = (const float*)d_in[14];
    const float* g2  = (const float*)d_in[15];
    const float* b2  = (const float*)d_in[16];
    float* out = (float*)d_out;

    size_t off = 0;
    auto carve = [&](size_t bytes) {
        void* p = (char*)d_ws + off;
        off += (bytes + 255) & ~(size_t)255;
        return p;
    };
    unsigned short* wqT  = (unsigned short*)carve((size_t)Dn * Dn * 2);
    unsigned short* wkT  = (unsigned short*)carve((size_t)Dn * KVHn * HDn * 2);
    unsigned short* wvT  = (unsigned short*)carve((size_t)Dn * KVHn * HDn * 2);
    unsigned short* woT  = (unsigned short*)carve((size_t)Dn * Dn * 2);
    unsigned short* wf1T = (unsigned short*)carve((size_t)Dn * DFFn * 2);
    unsigned short* wf2T = (unsigned short*)carve((size_t)DFFn * Dn * 2);
    unsigned short* hbf  = (unsigned short*)carve((size_t)Mrows * Dn * 2);
    float*          qf   = (float*)carve((size_t)Mrows * Dn * 4);
    float*          kf   = (float*)carve((size_t)Mrows * KVHn * HDn * 4);
    float*          vf   = (float*)carve((size_t)Mrows * KVHn * HDn * 4);
    unsigned short* qbf  = (unsigned short*)carve((size_t)Mrows * Dn * 2);
    unsigned short* kbf  = (unsigned short*)carve((size_t)Mrows * KVHn * HDn * 2);
    unsigned short* vtb  = (unsigned short*)carve((size_t)Mrows * KVHn * HDn * 2);
    unsigned short* abf  = (unsigned short*)carve((size_t)Mrows * Dn * 2);
    float*          x1   = (float*)carve((size_t)Mrows * Dn * 4);
    unsigned short* h2bf = (unsigned short*)carve((size_t)Mrows * Dn * 2);
    unsigned short* ffbf = (unsigned short*)carve((size_t)Mrows * DFFn * 2);

    const int KVD = KVHn * HDn;   // 256

    wconv_kernel<<<(Dn * Dn)   / 256, 256, 0, stream>>>(Wq,  wqT,  Dn,  Dn);
    wconv_kernel<<<(Dn * KVD)  / 256, 256, 0, stream>>>(Wk,  wkT,  Dn,  KVD);
    wconv_kernel<<<(Dn * KVD)  / 256, 256, 0, stream>>>(Wv,  wvT,  Dn,  KVD);
    wconv_kernel<<<(Dn * Dn)   / 256, 256, 0, stream>>>(Wo,  woT,  Dn,  Dn);
    wconv_kernel<<<(Dn * DFFn) / 256, 256, 0, stream>>>(Wf1, wf1T, Dn,  DFFn);
    wconv_kernel<<<(DFFn * Dn) / 256, 256, 0, stream>>>(Wf2, wf2T, DFFn, Dn);

    ln_kernel<<<Mrows, 256, 0, stream>>>(x, g1, b1, hbf, Dn);

    auto blocks = [](int M, int N) { return (M / 128) * (N / 64); };
    gemm_kernel<0><<<blocks(Mrows, Dn),  256, 0, stream>>>(hbf, wqT, bq, nullptr, qf, nullptr, Mrows, Dn,  Dn);
    gemm_kernel<0><<<blocks(Mrows, KVD), 256, 0, stream>>>(hbf, wkT, bk, nullptr, kf, nullptr, Mrows, KVD, Dn);
    gemm_kernel<0><<<blocks(Mrows, KVD), 256, 0, stream>>>(hbf, wvT, bv, nullptr, vf, nullptr, Mrows, KVD, Dn);

    rope_kernel<<<(Bn * Tn * Hn   * 32) / 256, 256, 0, stream>>>(qf, qbf, Hn);
    rope_kernel<<<(Bn * Tn * KVHn * 32) / 256, 256, 0, stream>>>(kf, kbf, KVHn);
    vtrans_kernel<<<(Bn * KVHn * HDn * Tn) / 256, 256, 0, stream>>>(vf, vtb);

    attn_kernel<<<Bn * Hn * (Tn / 16), 32, 0, stream>>>(qbf, kbf, vtb, abf);

    gemm_kernel<1><<<blocks(Mrows, Dn), 256, 0, stream>>>(abf, woT, bo, x, x1, nullptr, Mrows, Dn, Dn);

    ln_kernel<<<Mrows, 256, 0, stream>>>(x1, g2, b2, h2bf, Dn);

    gemm_kernel<2><<<blocks(Mrows, DFFn), 256, 0, stream>>>(h2bf, wf1T, bf1, nullptr, nullptr, ffbf, Mrows, DFFn, Dn);
    gemm_kernel<1><<<blocks(Mrows, Dn),   256, 0, stream>>>(ffbf, wf2T, bf2, x1, out, nullptr, Mrows, Dn, DFFn);
}
